// UncertaintyGuidedAttention_50603304681522
// MI455X (gfx1250) — compile-verified, matrix operand order checked by reading, not run on hardware
//
#include <hip/hip_runtime.h>
#include <hip/hip_bf16.h>

// ---------------------------------------------------------------------------
// UncertaintyGuidedAttention on gfx1250 (CDNA5, wave32, WMMA 16x16x32 f16)
//   B=2, C=256, H=W=48 -> N=2304 tokens, 8 heads, head_dim=32
// Pipeline:
//   K0: one-shot f32->f16 conversion of Wq/Wk/Wv/Wo into workspace
//   K1: gate + fused Q/K/V GEMMs, 32-token blocks, A-fragment reused for 2
//       token sub-tiles (2 WMMAs per weight fragment). SCALE folded into Q.
//   K2: flash attention, one wave per (b,h,32-query tile): 4 score WMMAs +
//       4 PV WMMAs per 32-key step, online softmax, (1-u[key]) score mask.
//   K3: output projection GEMM + bias + residual, 32-token blocks.
// ---------------------------------------------------------------------------

typedef __attribute__((ext_vector_type(16))) _Float16 v16h;
typedef __attribute__((ext_vector_type(8)))  float    v8f;

#define CCH   256
#define NTOK  2304
#define NH    8
#define HDIM  32
#define SCALE_F 0.17677669529663687f  // 1/sqrt(32)

__device__ __forceinline__ v8f wmma16(v16h a, v16h b, v8f c) {
  // D = A(16x32 f16) * B(32x16 f16) + C(16x16 f32)
  return __builtin_amdgcn_wmma_f32_16x16x32_f16(
      false, a, false, b, (short)0, c, false, false);
}

// A/B fragment, 16-bit elements: lane holds 16 halfs; element e in [0,8) maps
// to K = (lane>>4)*8 + e, element e in [8,16) maps to K = 16 + (lane>>4)*8+e-8
// -> two contiguous 8-half (16B) runs from a K-contiguous pointer.
__device__ __forceinline__ v16h frag_h(const _Float16* __restrict__ p, int h8) {
  v16h r;
#pragma unroll
  for (int i = 0; i < 8; ++i) r[i] = p[h8 + i];
#pragma unroll
  for (int i = 0; i < 8; ++i) r[8 + i] = p[16 + h8 + i];
  return r;
}

// reductions across the 16 lanes of a half-wave (xor masks < 16 never cross)
__device__ __forceinline__ float redmax16(float v) {
#pragma unroll
  for (int m = 1; m < 16; m <<= 1) v = fmaxf(v, __shfl_xor(v, m, 32));
  return v;
}
__device__ __forceinline__ float redsum16(float v) {
#pragma unroll
  for (int m = 1; m < 16; m <<= 1) v += __shfl_xor(v, m, 32);
  return v;
}

// ---------------------------------------------------------------------------
// Kernel 0: convert the four CxC weight matrices to f16 (row-major preserved).
// ---------------------------------------------------------------------------
__global__ void __launch_bounds__(256)
cvt_w_kernel(const float* __restrict__ wq, const float* __restrict__ wk,
             const float* __restrict__ wv, const float* __restrict__ wo,
             _Float16* __restrict__ oq, _Float16* __restrict__ ok,
             _Float16* __restrict__ ov, _Float16* __restrict__ oo) {
  const int i = blockIdx.x * 256 + threadIdx.x;  // grid = 256 -> CCH*CCH ids
  oq[i] = (_Float16)wq[i];
  ok[i] = (_Float16)wk[i];
  ov[i] = (_Float16)wv[i];
  oo[i] = (_Float16)wo[i];
}

// ---------------------------------------------------------------------------
// Kernel 1: uncertainty gate + Q/K/V projections.
// grid = (NTOK/32, B), block = 256 (8 waves). Block: 32 tokens, all 256 out
// channels for q,k,v -> 48 (mat,otile) combos, 6 per wave, each covering two
// 16-token sub-tiles with one shared weight A-fragment per K-step.
// q,k written [b,h,n,hd] (token-major, q pre-scaled), v written [b,h,hd,n].
// ---------------------------------------------------------------------------
__global__ void __launch_bounds__(256)
qkv_kernel(const float* __restrict__ x,  const float* __restrict__ u,
           const _Float16* __restrict__ wq16, const float* __restrict__ bq,
           const _Float16* __restrict__ wk16, const float* __restrict__ bk,
           const _Float16* __restrict__ wv16, const float* __restrict__ bv,
           const float* __restrict__ Wg, const float* __restrict__ bg,
           _Float16* __restrict__ qws, _Float16* __restrict__ kws,
           _Float16* __restrict__ vws) {
  __shared__ _Float16 ldsG[32 * CCH];  // gated x, [n][c] (c contiguous)
  __shared__ _Float16 ldsX[32 * CCH];  // raw   x, [n][c]
  const int t = threadIdx.x;
  const int b = blockIdx.y;
  const int n0 = blockIdx.x * 32;

  {  // stage: thread t owns channel c = t for all 32 tokens
    const float* xb = x + ((size_t)b * CCH + t) * NTOK + n0;
    const float* ub = u + (size_t)b * NTOK + n0;
    const float wg = Wg[t], bgc = bg[t];
#pragma unroll
    for (int n = 0; n < 32; ++n) {
      float xv = xb[n];
      float gate = 1.0f - (wg * ub[n] + bgc);
      ldsX[n * CCH + t] = (_Float16)xv;
      ldsG[n * CCH + t] = (_Float16)(xv * gate);
    }
  }
  __syncthreads();

  const int wave = t >> 5, lane = t & 31;
  const int col = lane & 15, h8 = (lane >> 4) * 8;

#pragma unroll 1
  for (int i = 0; i < 6; ++i) {
    const int tid = wave + 8 * i;          // 0..47 (wave-uniform)
    const int mat = tid >> 4;              // 0=q 1=k 2=v
    const int obase = (tid & 15) * 16;
    const _Float16* W = (mat == 0) ? wq16 : ((mat == 1) ? wk16 : wv16);
    const float* bias  = (mat == 0) ? bq : ((mat == 1) ? bk : bv);
    const _Float16* src = (mat == 2) ? ldsX : ldsG;
    const _Float16* wrow = W + (size_t)(obase + col) * CCH;  // A row = lane&15

    v8f acc0 = {}, acc1 = {};
#pragma unroll
    for (int kk = 0; kk < CCH; kk += 32) {
      __builtin_prefetch(wrow + kk + 32, 0, 3);
      v16h a  = frag_h(wrow + kk, h8);                    // weights 16o x 32c
      v16h b0 = frag_h(src + col * CCH + kk, h8);         // tokens n0..n0+15
      v16h b1 = frag_h(src + (col + 16) * CCH + kk, h8);  // tokens +16..+31
      acc0 = wmma16(a, b0, acc0);
      acc1 = wmma16(a, b1, acc1);
    }
    const float post = (mat == 0) ? SCALE_F : 1.0f;  // fold 1/sqrt(hd) into q
#pragma unroll
    for (int j = 0; j < 8; ++j) {
      const int o = obase + j + 8 * (lane >> 4);   // D row
      const float bv0 = bias[o];
      const float valA = (acc0[j] + bv0) * post;
      const float valB = (acc1[j] + bv0) * post;
      const int nA = n0 + col, nB = n0 + 16 + col;
      const int h = o >> 5, d = o & 31;
      if (mat == 2) {
        _Float16* dst = vws + (((size_t)b * NH + h) * HDIM + d) * NTOK;
        dst[nA] = (_Float16)valA;
        dst[nB] = (_Float16)valB;
      } else {
        _Float16* dst = ((mat == 0) ? qws : kws) +
                        ((size_t)b * NH + h) * NTOK * HDIM + d;
        dst[(size_t)nA * HDIM] = (_Float16)valA;
        dst[(size_t)nB * HDIM] = (_Float16)valB;
      }
    }
  }
}

// ---------------------------------------------------------------------------
// Kernel 2 helpers: online-softmax update + epilogue for one 16-query tile.
// ---------------------------------------------------------------------------
__device__ __forceinline__ void softmax_update(
    const v8f& sA, const v8f& sB, float f0, float f1, float* mrun,
    float* lrun, v8f& accA, v8f& accB, _Float16* pbuf, int rowbase, int lane) {
  const int col = lane & 15;
#pragma unroll
  for (int j = 0; j < 8; ++j) {
    const float a = sA[j] * f0, c = sB[j] * f1;
    const float tmax = redmax16(fmaxf(a, c));
    const float mnew = fmaxf(mrun[j], tmax);
    const float alpha = __expf(mrun[j] - mnew);
    const float p0 = __expf(a - mnew);
    const float p1 = __expf(c - mnew);
    lrun[j] = lrun[j] * alpha + redsum16(p0 + p1);
    mrun[j] = mnew;
    accA[j] *= alpha;
    accB[j] *= alpha;
    const int row = rowbase + j + 8 * (lane >> 4);
    pbuf[row * 32 + col]      = (_Float16)p0;
    pbuf[row * 32 + col + 16] = (_Float16)p1;
  }
}

// ---------------------------------------------------------------------------
// Kernel 2: flash attention. grid = (NTOK/32, NH, B), block = 32 (one wave).
// One wave owns a 32-query tile (two A fragments); per 32-key step the K and
// V fragments are loaded once and feed both query tiles: 8 WMMAs per step.
// ---------------------------------------------------------------------------
__global__ void __launch_bounds__(32)
attn_kernel(const _Float16* __restrict__ qws, const _Float16* __restrict__ kws,
            const _Float16* __restrict__ vws, const float* __restrict__ u,
            _Float16* __restrict__ ows) {
  __shared__ _Float16 pbuf[32 * 32];  // probs [query][key] for C->A transpose
  const int lane = threadIdx.x;
  const int col = lane & 15, h8 = (lane >> 4) * 8;
  const int q0 = blockIdx.x * 32, h = blockIdx.y, b = blockIdx.z;
  const size_t bh = (size_t)b * NH + h;
  const _Float16* Q = qws + bh * NTOK * HDIM;   // [n][d], pre-scaled
  const _Float16* K = kws + bh * NTOK * HDIM;   // [n][d]
  const _Float16* V = vws + bh * HDIM * NTOK;   // [d][n]
  const float* ub = u + (size_t)b * NTOK;

  const v16h qa0 = frag_h(Q + (size_t)(q0 + col) * HDIM, h8);       // q tile 0
  const v16h qa1 = frag_h(Q + (size_t)(q0 + 16 + col) * HDIM, h8);  // q tile 1

  v8f acc00 = {}, acc01 = {}, acc10 = {}, acc11 = {};  // [qtile][d lo/hi]
  float m0[8], l0[8], m1[8], l1[8];
#pragma unroll
  for (int j = 0; j < 8; ++j) {
    m0[j] = -1e30f; l0[j] = 0.0f;
    m1[j] = -1e30f; l1[j] = 0.0f;
  }

#pragma unroll 1
  for (int kk = 0; kk < NTOK; kk += 32) {
    v16h kb0 = frag_h(K + (size_t)(kk + col) * HDIM, h8);       // 32d x 16k
    v16h kb1 = frag_h(K + (size_t)(kk + 16 + col) * HDIM, h8);
    v8f z = {};
    v8f s00 = wmma16(qa0, kb0, z);
    v8f s01 = wmma16(qa0, kb1, z);
    v8f s10 = wmma16(qa1, kb0, z);
    v8f s11 = wmma16(qa1, kb1, z);
    const float f0 = 1.0f - ub[kk + col];        // key-axis uncertainty mask
    const float f1 = 1.0f - ub[kk + 16 + col];
    softmax_update(s00, s01, f0, f1, m0, l0, acc00, acc01, pbuf, 0, lane);
    softmax_update(s10, s11, f0, f1, m1, l1, acc10, acc11, pbuf, 16, lane);
    __syncthreads();
    v16h pa0 = frag_h(pbuf + col * 32, h8);              // q tile 0: 16q x 32k
    v16h pa1 = frag_h(pbuf + (col + 16) * 32, h8);       // q tile 1
    v16h vb0 = frag_h(V + (size_t)col * NTOK + kk, h8);  // 32k x 16d
    v16h vb1 = frag_h(V + (size_t)(col + 16) * NTOK + kk, h8);
    acc00 = wmma16(pa0, vb0, acc00);
    acc01 = wmma16(pa0, vb1, acc01);
    acc10 = wmma16(pa1, vb0, acc10);
    acc11 = wmma16(pa1, vb1, acc11);
    __syncthreads();
  }

#pragma unroll
  for (int j = 0; j < 8; ++j) {
    const int r = j + 8 * (lane >> 4);
    const float inv0 = 1.0f / l0[j], inv1 = 1.0f / l1[j];
    const size_t baseA = ((size_t)b * NTOK + q0 + r) * CCH + h * HDIM;
    const size_t baseB = ((size_t)b * NTOK + q0 + 16 + r) * CCH + h * HDIM;
    ows[baseA + col]      = (_Float16)(acc00[j] * inv0);  // [b][n][c]
    ows[baseA + col + 16] = (_Float16)(acc01[j] * inv0);
    ows[baseB + col]      = (_Float16)(acc10[j] * inv1);
    ows[baseB + col + 16] = (_Float16)(acc11[j] * inv1);
  }
}

// ---------------------------------------------------------------------------
// Kernel 3: out = Wo @ attn_out + bo + x. grid = (NTOK/32, B), block = 256.
// Each wave: 2 output-channel tiles x 2 token sub-tiles (shared A fragment).
// ---------------------------------------------------------------------------
__global__ void __launch_bounds__(256)
proj_kernel(const _Float16* __restrict__ ows, const _Float16* __restrict__ wo16,
            const float* __restrict__ bo, const float* __restrict__ x,
            float* __restrict__ out) {
  const int t = threadIdx.x;
  const int b = blockIdx.y;
  const int n0 = blockIdx.x * 32;
  const int wave = t >> 5, lane = t & 31;
  const int col = lane & 15, h8 = (lane >> 4) * 8;
  const _Float16* src0 = ows + ((size_t)b * NTOK + n0 + col) * CCH;
  const _Float16* src1 = src0 + (size_t)16 * CCH;

#pragma unroll 1
  for (int i = 0; i < 2; ++i) {
    const int obase = (wave + 8 * i) * 16;
    const _Float16* wrow = wo16 + (size_t)(obase + col) * CCH;
    v8f acc0 = {}, acc1 = {};
#pragma unroll
    for (int kk = 0; kk < CCH; kk += 32) {
      __builtin_prefetch(wrow + kk + 32, 0, 3);
      v16h a  = frag_h(wrow + kk, h8);  // Wo 16o x 32c
      v16h b0 = frag_h(src0 + kk, h8);  // attn_out 32c x 16n
      v16h b1 = frag_h(src1 + kk, h8);
      acc0 = wmma16(a, b0, acc0);
      acc1 = wmma16(a, b1, acc1);
    }
#pragma unroll
    for (int j = 0; j < 8; ++j) {
      const int o = obase + j + 8 * (lane >> 4);
      const float bv0 = bo[o];
      const size_t idxA = ((size_t)b * CCH + o) * NTOK + n0 + col;
      const size_t idxB = idxA + 16;
      out[idxA] = acc0[j] + bv0 + x[idxA];
      out[idxB] = acc1[j] + bv0 + x[idxB];
    }
  }
}

// ---------------------------------------------------------------------------
extern "C" void kernel_launch(void* const* d_in, const int* in_sizes, int n_in,
                              void* d_out, int out_size, void* d_ws,
                              size_t ws_size, hipStream_t stream) {
  const float* x  = (const float*)d_in[0];
  const float* u  = (const float*)d_in[1];
  const float* Wq = (const float*)d_in[2];  const float* bq = (const float*)d_in[3];
  const float* Wk = (const float*)d_in[4];  const float* bk = (const float*)d_in[5];
  const float* Wv = (const float*)d_in[6];  const float* bv = (const float*)d_in[7];
  const float* Wg = (const float*)d_in[8];  const float* bg = (const float*)d_in[9];
  const float* Wo = (const float*)d_in[10]; const float* bo = (const float*)d_in[11];

  const int B = in_sizes[0] / (CCH * NTOK);  // = 2
  const size_t per = (size_t)B * NH * NTOK * HDIM;  // elems per q/k/v/o buffer
  const size_t wsz = (size_t)CCH * CCH;             // elems per f16 weight

  _Float16* qws  = (_Float16*)d_ws;  // [b,h,n,hd] (pre-scaled by 1/sqrt(hd))
  _Float16* kws  = qws + per;        // [b,h,n,hd]
  _Float16* vws  = kws + per;        // [b,h,hd,n]
  _Float16* ows  = vws + per;        // [b,n,c]
  _Float16* wq16 = ows + per;
  _Float16* wk16 = wq16 + wsz;
  _Float16* wv16 = wk16 + wsz;
  _Float16* wo16 = wv16 + wsz;

  cvt_w_kernel<<<dim3(CCH * CCH / 256), 256, 0, stream>>>(
      Wq, Wk, Wv, Wo, wq16, wk16, wv16, wo16);
  qkv_kernel<<<dim3(NTOK / 32, B), 256, 0, stream>>>(
      x, u, wq16, bq, wk16, bk, wv16, bv, Wg, bg, qws, kws, vws);
  attn_kernel<<<dim3(NTOK / 32, NH, B), 32, 0, stream>>>(qws, kws, vws, u, ows);
  proj_kernel<<<dim3(NTOK / 32, B), 256, 0, stream>>>(ows, wo16, bo, x,
                                                      (float*)d_out);
}
